// PNA_72009421684762
// MI455X (gfx1250) — matrix operand order repaired and checked by reading, not compiled, not run on hardware
//
#include <hip/hip_runtime.h>
#include <hip/hip_bf16.h>
#include <math.h>

// ---------------- problem constants ----------------
static constexpr int Nn   = 50000;
static constexpr int Ee   = 800000;
static constexpr int XD   = 64;
static constexpr int EA   = 16;
static constexpr int HID  = 64;
static constexpr int NL   = 3;
static constexpr int NC   = 10;
static constexpr int NG   = 64;
static constexpr int K768 = 768;        // 4 aggregators * 192
static constexpr int KT   = 24;         // 768/32 k-tiles
static constexpr int TILES = Nn / 16;   // 3125 row tiles (exact)
static constexpr int RECS_PER_LAYER = KT * 3 * 4;          // 288 B-records per layer
static constexpr int HALVES_PER_LAYER = RECS_PER_LAYER * 512; // 147456
static constexpr int PACK_TOT = NL * HALVES_PER_LAYER;

typedef __attribute__((ext_vector_type(16))) __bf16 v16bf;
typedef __attribute__((ext_vector_type(8)))  float  v8f;

__device__ __forceinline__ unsigned short f2bf(float f) { // RNE f32 -> bf16
    unsigned u = __float_as_uint(f);
    unsigned r = u + 0x7FFFu + ((u >> 16) & 1u);
    return (unsigned short)(r >> 16);
}

union V16 { v16bf v; uint4 q[2]; };

// ---------------- one-time prep kernels ----------------

// Pack conv_w [NL][2304][64] f32 -> per-lane WMMA-B bf16 records.
__global__ void pack_w_kernel(const float* __restrict__ conv_w,
                              unsigned short* __restrict__ wpack) {
    int gid = blockIdx.x * blockDim.x + threadIdx.x;
    if (gid >= PACK_TOT) return;
    int i    = gid & 15;
    int lane = (gid >> 4) & 31;
    int rec  = gid >> 9;
    int layer = rec / RECS_PER_LAYER;
    int r2    = rec % RECS_PER_LAYER;
    int kt    = r2 / 12;
    int r3    = r2 % 12;
    int chunk = r3 >> 2;
    int ntile = r3 & 3;
    int h  = lane >> 4;
    int k  = kt * 32 + (i < 8 ? 8 * h + i : 16 + 8 * h + (i - 8));     // B layout: lanes 0-15 K=0..7,16..23; 16-31 K=8..15,24..31
    int n  = ntile * 16 + (lane & 15);
    int row = chunk * K768 + k;
    float v = conv_w[((size_t)layer * 2304 + row) * 64 + n];
    wpack[(size_t)rec * 512 + lane * 16 + i] = f2bf(v);
}

// x_enc = x @ node_w + node_b
__global__ void node_enc_kernel(const float* __restrict__ x,
                                const float* __restrict__ w,
                                const float* __restrict__ b,
                                float* __restrict__ xo) {
    int gid = blockIdx.x * blockDim.x + threadIdx.x;
    if (gid >= Nn * HID) return;
    int n = gid >> 6, c = gid & 63;
    const float* xr = x + (size_t)n * XD;
    float acc = b[c];
    #pragma unroll 8
    for (int k = 0; k < XD; ++k) acc = fmaf(xr[k], w[k * HID + c], acc);
    xo[gid] = acc;
}

// ea_enc = edge_attr @ edge_w + edge_b   (layer-invariant: computed once)
__global__ void edge_enc_kernel(const float* __restrict__ ea,
                                const float* __restrict__ w,
                                const float* __restrict__ b,
                                float* __restrict__ eo) {
    int gid = blockIdx.x * blockDim.x + threadIdx.x;
    if (gid >= Ee * HID) return;
    int e = gid >> 6, c = gid & 63;
    const float* er = ea + (size_t)e * EA;
    float acc = b[c];
    #pragma unroll
    for (int k = 0; k < EA; ++k) acc = fmaf(er[k], w[k * HID + c], acc);
    eo[gid] = acc;
}

__global__ void deg_count_kernel(const int* __restrict__ dst, int* __restrict__ degi) {
    int e = blockIdx.x * blockDim.x + threadIdx.x;
    if (e >= Ee) return;
    atomicAdd(&degi[dst[e]], 1);
}

// Single-block exclusive scan over degi -> offsets[0..Nn], cursor copy.
__global__ void scan_kernel(const int* __restrict__ degi,
                            int* __restrict__ offsets,
                            int* __restrict__ cursor) {
    __shared__ int buf[1024];
    int tid = threadIdx.x;
    int carry = 0;
    for (int base = 0; base < Nn; base += 1024) {
        int i = base + tid;
        int v = (i < Nn) ? degi[i] : 0;
        buf[tid] = v;
        __syncthreads();
        #pragma unroll
        for (int offd = 1; offd < 1024; offd <<= 1) {
            int t = (tid >= offd) ? buf[tid - offd] : 0;
            __syncthreads();
            buf[tid] += t;
            __syncthreads();
        }
        int excl = buf[tid] - v;
        if (i < Nn) { offsets[i] = carry + excl; cursor[i] = carry + excl; }
        int total = buf[1023];
        __syncthreads();
        carry += total;
    }
    if (tid == 0) offsets[Nn] = carry;
}

__global__ void fill_elist_kernel(const int* __restrict__ dst,
                                  int* __restrict__ cursor,
                                  int* __restrict__ elist) {
    int e = blockIdx.x * blockDim.x + threadIdx.x;
    if (e >= Ee) return;
    int pos = atomicAdd(&cursor[dst[e]], 1);
    elist[pos] = e;
}

__global__ void deg_prep_kernel(const int* __restrict__ degi,
                                float* __restrict__ logdeg,
                                float* __restrict__ avgAcc) {
    int n = blockIdx.x * blockDim.x + threadIdx.x;
    if (n >= Nn) return;
    float dg = (float)degi[n];
    logdeg[n] = logf(fmaxf(dg, 1.0f) + 1.0f);
    atomicAdd(avgAcc, logf(dg + 1.0f));
}

__global__ void amp_att_kernel(const float* __restrict__ logdeg,
                               const float* __restrict__ avgAcc,
                               float* __restrict__ amp,
                               float* __restrict__ att) {
    int n = blockIdx.x * blockDim.x + threadIdx.x;
    if (n >= Nn) return;
    float avg = avgAcc[0] * (1.0f / (float)Nn);
    float ld  = logdeg[n];
    amp[n] = ld / avg;
    att[n] = avg / ld;
}

// ---------------- per-layer kernels ----------------

// Gather-based segment reduction + feature finalize (no atomics).
// Block = 192 threads (one per message feature), grid = Nn (one block per node).
__global__ void agg_feat_kernel(const float* __restrict__ x,
                                const float* __restrict__ ea_enc,
                                const int* __restrict__ src,
                                const int* __restrict__ offsets,
                                const int* __restrict__ elist,
                                unsigned short* __restrict__ feat) {
    __shared__ int eidsh[256];
    __shared__ int srcsh[256];
    int n = blockIdx.x;
    int j = threadIdx.x;
    int beg = offsets[n], end = offsets[n + 1];
    int d = end - beg;
    float dgf = (float)d;
    float inv = 1.0f / fmaxf(dgf, 1.0f);

    float s, q, mx, mn;
    float xv = 0.0f;
    if (j < 64) xv = x[(size_t)n * HID + j];   // constant across this node's edges
    s = 0.0f; q = 0.0f; mx = -3.402823466e38f; mn = 3.402823466e38f;

    for (int c0 = beg; c0 < end; c0 += 256) {
        int m = min(256, end - c0);
        for (int t = j; t < m; t += 192) {
            int ee = elist[c0 + t];
            eidsh[t] = ee;
            srcsh[t] = src[ee];
        }
        __syncthreads();
        if (j >= 64 && j < 128) {
            const float* base = x + (j - 64);
            for (int t = 0; t < m; ++t) {
                float v = base[(size_t)srcsh[t] * HID];
                s += v; q += v * v; mx = fmaxf(mx, v); mn = fminf(mn, v);
            }
        } else if (j >= 128) {
            const float* base = ea_enc + (j - 128);
            for (int t = 0; t < m; ++t) {
                float v = base[(size_t)eidsh[t] * HID];
                s += v; q += v * v; mx = fmaxf(mx, v); mn = fminf(mn, v);
            }
        }
        __syncthreads();
    }
    if (j < 64) {  // closed form: every message carries x[dst][j]
        s = dgf * xv; q = dgf * xv * xv; mx = xv; mn = xv;
    }
    float mean = s * inv;
    float var  = q * inv - mean * mean;
    float stdv = sqrtf(fmaxf(var, 0.0f) + 1e-5f);
    if (d == 0) { mx = 0.0f; mn = 0.0f; }
    unsigned short* f = feat + (size_t)n * K768;
    f[j]       = f2bf(mean);
    f[192 + j] = f2bf(mn);
    f[384 + j] = f2bf(mx);
    f[576 + j] = f2bf(stdv);
}

// WMMA GEMM: h[N,64] = feat@W0 + amp*(feat@W1) + att*(feat@W2) + bias.
// blockDim=128 (4 waves); each wave owns ONE 16-row tile: 12 f32 accumulators
// (96 VGPRs) -- spill-free under the 256-VGPR direct-addressing window
// (round-1 codegen verified: clause-grouped b128 loads, no scratch traffic).
__global__ void gemm_conv_kernel(const unsigned short* __restrict__ feat,
                                 const unsigned short* __restrict__ wpack,
                                 const float* __restrict__ amp,
                                 const float* __restrict__ att,
                                 const float* __restrict__ convb,
                                 float* __restrict__ hout) {
    int wave = threadIdx.x >> 5;
    int lane = threadIdx.x & 31;
    int tile = blockIdx.x * 4 + wave;
    if (tile >= TILES) return;                 // wave-uniform: EXEC stays all-ones
    int r  = lane & 15;
    int hh = lane >> 4;

    const unsigned short* arow = feat + (size_t)(tile * 16 + r) * K768;

    v8f zero = {0.f, 0.f, 0.f, 0.f, 0.f, 0.f, 0.f, 0.f};
    v8f acc[3][4];
    #pragma unroll
    for (int c = 0; c < 3; ++c)
        #pragma unroll
        for (int t = 0; t < 4; ++t) acc[c][t] = zero;

    for (int kt = 0; kt < KT; ++kt) {
        V16 a;
        a.q[0] = *(const uint4*)(arow + kt * 32 + 8 * hh);
        a.q[1] = *(const uint4*)(arow + kt * 32 + 16 + 8 * hh);
        const uint4* bbase = (const uint4*)wpack + ((size_t)kt * 12) * 64 + lane * 2;
        #pragma unroll
        for (int c = 0; c < 3; ++c) {
            #pragma unroll
            for (int t = 0; t < 4; ++t) {
                V16 b;
                const uint4* qp = bbase + (size_t)(c * 4 + t) * 64;
                b.q[0] = qp[0];
                b.q[1] = qp[1];
                acc[c][t] = __builtin_amdgcn_wmma_f32_16x16x32_bf16(
                    false, a.v, false, b.v, (short)0, acc[c][t], false, false);
            }
        }
    }

    // epilogue: fold degree scalers + bias; C/D layout: elem(j) = (M=j+8*hh, N=r)
    int row0 = tile * 16 + 8 * hh;
    float av[8], tv[8];
    #pragma unroll
    for (int j = 0; j < 8; ++j) { av[j] = amp[row0 + j]; tv[j] = att[row0 + j]; }
    #pragma unroll
    for (int t = 0; t < 4; ++t) {
        float bcol = convb[t * 16 + r];
        #pragma unroll
        for (int j = 0; j < 8; ++j) {
            float v = acc[0][t][j] + av[j] * acc[1][t][j] + tv[j] * acc[2][t][j] + bcol;
            hout[(size_t)(row0 + j) * HID + t * 16 + r] = v;
        }
    }
}

// BatchNorm stats: per-channel sum / sumsq. blockDim = 64.
__global__ void bn_stats_kernel(const float* __restrict__ hmat,
                                float* __restrict__ bnsum,
                                float* __restrict__ bnsq) {
    int c  = threadIdx.x;
    int r0 = blockIdx.x * 256;
    int r1 = min(r0 + 256, Nn);
    float s = 0.f, q = 0.f;
    for (int r = r0; r < r1; ++r) {
        float v = hmat[(size_t)r * HID + c];
        s += v; q += v * v;
    }
    atomicAdd(&bnsum[c], s);
    atomicAdd(&bnsq[c],  q);
}

// x = relu(BN(h)) + x
__global__ void bn_apply_kernel(const float* __restrict__ hmat,
                                const float* __restrict__ bnsum,
                                const float* __restrict__ bnsq,
                                const float* __restrict__ g,
                                const float* __restrict__ b,
                                float* __restrict__ xio) {
    int gid = blockIdx.x * blockDim.x + threadIdx.x;
    if (gid >= Nn * HID) return;
    int c = gid & 63;
    float mu  = bnsum[c] * (1.0f / (float)Nn);
    float var = bnsq[c]  * (1.0f / (float)Nn) - mu * mu;
    float v = (hmat[gid] - mu) * rsqrtf(var + 1e-5f) * g[c] + b[c];
    xio[gid] = fmaxf(v, 0.0f) + xio[gid];
}

// ---------------- head ----------------
__global__ void pool_kernel(const float* __restrict__ x,
                            const int* __restrict__ batch,
                            float* __restrict__ gsum,
                            float* __restrict__ gcnt) {
    int gid = blockIdx.x * blockDim.x + threadIdx.x;
    if (gid >= Nn * HID) return;
    int n = gid >> 6, c = gid & 63;
    int bidx = batch[n];
    atomicAdd(&gsum[(size_t)bidx * HID + c], x[gid]);
    if (c == 0) atomicAdd(&gcnt[bidx], 1.0f);
}

__global__ void fc_head_kernel(const float* __restrict__ gsum,
                               const float* __restrict__ gcnt,
                               const float* __restrict__ w1, const float* __restrict__ b1,
                               const float* __restrict__ w2, const float* __restrict__ b2,
                               const float* __restrict__ w3, const float* __restrict__ b3,
                               float* __restrict__ out) {
    __shared__ float gin[64], h1[32], h2[16];
    int g = blockIdx.x, t = threadIdx.x;
    float cnt = fmaxf(gcnt[g], 1.0f);
    gin[t] = gsum[(size_t)g * HID + t] / cnt;
    __syncthreads();
    if (t < 32) {
        float a = b1[t];
        #pragma unroll 8
        for (int k = 0; k < 64; ++k) a = fmaf(gin[k], w1[k * 32 + t], a);
        h1[t] = fmaxf(a, 0.0f);
    }
    __syncthreads();
    if (t < 16) {
        float a = b2[t];
        #pragma unroll
        for (int k = 0; k < 32; ++k) a = fmaf(h1[k], w2[k * 16 + t], a);
        h2[t] = fmaxf(a, 0.0f);
    }
    __syncthreads();
    if (t < 10) {
        float a = b3[t];
        #pragma unroll
        for (int k = 0; k < 16; ++k) a = fmaf(h2[k], w3[k * 10 + t], a);
        out[g * NC + t] = a;
    }
}

// ---------------- host launch ----------------
extern "C" void kernel_launch(void* const* d_in, const int* in_sizes, int n_in,
                              void* d_out, int out_size, void* d_ws, size_t ws_size,
                              hipStream_t stream) {
    (void)in_sizes; (void)n_in; (void)out_size; (void)ws_size;
    const float* x      = (const float*)d_in[0];
    const int*   ei     = (const int*)  d_in[1];
    const int*   batch  = (const int*)  d_in[2];
    const float* eattr  = (const float*)d_in[3];
    const float* node_w = (const float*)d_in[4];
    const float* node_b = (const float*)d_in[5];
    const float* edge_w = (const float*)d_in[6];
    const float* edge_b = (const float*)d_in[7];
    const float* conv_w = (const float*)d_in[8];
    const float* conv_b = (const float*)d_in[9];
    const float* bn_g   = (const float*)d_in[10];
    const float* bn_b   = (const float*)d_in[11];
    const float* fc1_w  = (const float*)d_in[12];
    const float* fc1_b  = (const float*)d_in[13];
    const float* fc2_w  = (const float*)d_in[14];
    const float* fc2_b  = (const float*)d_in[15];
    const float* fc3_w  = (const float*)d_in[16];
    const float* fc3_b  = (const float*)d_in[17];
    float* out = (float*)d_out;

    const int* srcp = ei;
    const int* dstp = ei + Ee;

    char* ws = (char*)d_ws;
    size_t off = 0;
    auto take = [&](size_t bytes) -> char* {
        char* p = ws + off;
        off += (bytes + 255) & ~(size_t)255;
        return p;
    };
    const size_t SZ_X = (size_t)Nn * HID * 4;
    float*          x_enc   = (float*)take(SZ_X);
    float*          h_out   = (float*)take(SZ_X);
    float*          ea_enc  = (float*)take((size_t)Ee * HID * 4);
    int*            degi    = (int*)take((size_t)Nn * 4);
    int*            offsets = (int*)take((size_t)(Nn + 1) * 4);
    int*            cursor  = (int*)take((size_t)Nn * 4);
    int*            elist   = (int*)take((size_t)Ee * 4);
    float*          logdeg  = (float*)take((size_t)Nn * 4);
    float*          amp     = (float*)take((size_t)Nn * 4);
    float*          att     = (float*)take((size_t)Nn * 4);
    float*          avgAcc  = (float*)take(256);
    unsigned short* feat    = (unsigned short*)take((size_t)Nn * K768 * 2);
    unsigned short* wpack   = (unsigned short*)take((size_t)PACK_TOT * 2);
    float*          bnsum   = (float*)take(256);
    float*          bnsq    = (float*)take(256);
    float*          gsum    = (float*)take((size_t)NG * HID * 4);
    float*          gcnt    = (float*)take(256);

    // one-time prep
    pack_w_kernel<<<(PACK_TOT + 255) / 256, 256, 0, stream>>>(conv_w, wpack);
    node_enc_kernel<<<(Nn * HID + 255) / 256, 256, 0, stream>>>(x, node_w, node_b, x_enc);
    edge_enc_kernel<<<(Ee * HID + 255) / 256, 256, 0, stream>>>(eattr, edge_w, edge_b, ea_enc);

    hipMemsetAsync(degi, 0, (size_t)Nn * 4, stream);
    hipMemsetAsync(avgAcc, 0, 256, stream);
    deg_count_kernel<<<(Ee + 255) / 256, 256, 0, stream>>>(dstp, degi);
    scan_kernel<<<1, 1024, 0, stream>>>(degi, offsets, cursor);
    fill_elist_kernel<<<(Ee + 255) / 256, 256, 0, stream>>>(dstp, cursor, elist);
    deg_prep_kernel<<<(Nn + 255) / 256, 256, 0, stream>>>(degi, logdeg, avgAcc);
    amp_att_kernel<<<(Nn + 255) / 256, 256, 0, stream>>>(logdeg, avgAcc, amp, att);

    for (int l = 0; l < NL; ++l) {
        hipMemsetAsync(bnsum, 0, 256, stream);
        hipMemsetAsync(bnsq,  0, 256, stream);
        agg_feat_kernel<<<Nn, 192, 0, stream>>>(x_enc, ea_enc, srcp, offsets, elist, feat);
        gemm_conv_kernel<<<(TILES + 3) / 4, 128, 0, stream>>>(
            feat, wpack + (size_t)l * HALVES_PER_LAYER, amp, att, conv_b + l * HID, h_out);
        bn_stats_kernel<<<(Nn + 255) / 256, 64, 0, stream>>>(h_out, bnsum, bnsq);
        bn_apply_kernel<<<(Nn * HID + 255) / 256, 256, 0, stream>>>(
            h_out, bnsum, bnsq, bn_g + l * HID, bn_b + l * HID, x_enc);
    }

    hipMemsetAsync(gsum, 0, (size_t)NG * HID * 4, stream);
    hipMemsetAsync(gcnt, 0, 256, stream);
    pool_kernel<<<(Nn * HID + 255) / 256, 256, 0, stream>>>(x_enc, batch, gsum, gcnt);
    fc_head_kernel<<<NG, 64, 0, stream>>>(gsum, gcnt, fc1_w, fc1_b, fc2_w, fc2_b,
                                          fc3_w, fc3_b, out);
}
